// MPOLinear_55843164783339
// MI455X (gfx1250) — matrix operand order repaired and checked by reading, not compiled, not run on hardware
//
#include <hip/hip_runtime.h>
#include <hip/hip_bf16.h>
#include <stddef.h>

#define TOKENS 8192
#define IN_F   4096
#define OUT_F  4096

typedef __attribute__((ext_vector_type(16))) __bf16 v16bf;
typedef __attribute__((ext_vector_type(8)))  __bf16 v8bf;
typedef __attribute__((ext_vector_type(8)))  float  v8f;

__device__ __forceinline__ v16bf cat8(v8bf a, v8bf b) {
  return __builtin_shufflevector(a, b, 0,1,2,3,4,5,6,7,8,9,10,11,12,13,14,15);
}

// ---------------------------------------------------------------------------
// Kernel 1: T12[r1][o2*4096 + i2*256 + o3*16 + i3] =
//           sum_r2 core1[r1][o2][i2][r2] * core2[r2][o3][i3][0]
// block = (r1, o2i2) flattened (64*256 = 16384 blocks), thread = o3i3.
// core1 row is block-uniform (scalar loads); core2 reads are coalesced.
// ---------------------------------------------------------------------------
__global__ __launch_bounds__(256) void k_t12(const float* __restrict__ core1,
                                             const float* __restrict__ core2,
                                             float* __restrict__ t12) {
  const int b   = blockIdx.x;
  const int tid = threadIdx.x;          // o3*16 + i3
  const int r1  = b >> 8;
  const int c   = b & 255;              // o2*16 + i2
  const float* row = core1 + ((size_t)r1 * 256 + c) * 64;
  float acc = 0.0f;
#pragma unroll 8
  for (int r2 = 0; r2 < 64; ++r2)
    acc += row[r2] * core2[r2 * 256 + tid];
  t12[(size_t)r1 * 65536 + (size_t)c * 256 + tid] = acc;
}

// ---------------------------------------------------------------------------
// Kernel 2: W[o][i] = sum_r1 core0[o1*16+i1][r1] * T12[r1][inner],
// split into bf16 hi/lo planes. One block per inner-chunk (o2,i2 fixed,
// all o3,i3). T12 chunk [64 x 256] staged in LDS once, reused for all
// 256 (o1,i1) combos; core0 reads are uniform -> scalar path.
// ---------------------------------------------------------------------------
__global__ __launch_bounds__(256) void k_wsplit(const float* __restrict__ core0,
                                                const float* __restrict__ t12,
                                                __bf16* __restrict__ wh,
                                                __bf16* __restrict__ wl) {
  __shared__ float sT[64 * 256];        // 64 KB
  const int c   = blockIdx.x;           // o2*16 + i2
  const int tid = threadIdx.x;          // o3*16 + i3
  for (int r1 = 0; r1 < 64; ++r1)
    sT[r1 * 256 + tid] = t12[(size_t)r1 * 65536 + (size_t)c * 256 + tid];
  __syncthreads();

  const int o2 = c >> 4, i2 = c & 15;
  const int o3 = tid >> 4, i3 = tid & 15;

  for (int o1 = 0; o1 < 16; ++o1) {
    const int o = o1 * 256 + o2 * 16 + o3;
    for (int i1g = 0; i1g < 2; ++i1g) {
      float acc[8];
#pragma unroll
      for (int j = 0; j < 8; ++j) acc[j] = 0.0f;
      for (int r1 = 0; r1 < 64; ++r1) {
        const float t = sT[r1 * 256 + tid];
#pragma unroll
        for (int j = 0; j < 8; ++j)
          acc[j] += core0[(o1 * 16 + i1g * 8 + j) * 64 + r1] * t;  // uniform
      }
#pragma unroll
      for (int j = 0; j < 8; ++j) {
        const int i1 = i1g * 8 + j;
        const int i  = i1 * 256 + i2 * 16 + i3;
        const float  v = acc[j];
        const __bf16 h = (__bf16)v;
        wh[(size_t)o * IN_F + i] = h;
        wl[(size_t)o * IN_F + i] = (__bf16)(v - (float)h);
      }
    }
  }
}

// ---------------------------------------------------------------------------
// Kernel 3: y = x @ W^T via split-bf16 WMMA (hh + hl + lh ~ fp32 product).
// Block tile 128(M) x 128(N), K-chunks of 32; 8 waves, each 32x64 outputs
// (2x4 tiles of 16x16). Software-pipelined: next chunk's globals are
// prefetched into registers (ping-pong Stage structs) while the current
// chunk's WMMAs run out of LDS.
// ---------------------------------------------------------------------------
#define BM 128
#define BN 128
#define BK 32
#define LDSTR 40   // halfs per LDS row: 80 B, 16-B aligned, bank-spread

struct Stage {
  float4 x0, x1, x2, x3;   // 16 fp32 of x
  v8bf   wh0, wh1;         // 16 bf16 of Wh
  v8bf   wl0, wl1;         // 16 bf16 of Wl
};

__device__ __forceinline__ void stage_load(Stage& s, const float* gx,
                                           const __bf16* gwh, const __bf16* gwl,
                                           int kc) {
  s.x0  = *(const float4*)(gx + kc);
  s.x1  = *(const float4*)(gx + kc + 4);
  s.x2  = *(const float4*)(gx + kc + 8);
  s.x3  = *(const float4*)(gx + kc + 12);
  s.wh0 = *(const v8bf*)(gwh + kc);
  s.wh1 = *(const v8bf*)(gwh + kc + 8);
  s.wl0 = *(const v8bf*)(gwl + kc);
  s.wl1 = *(const v8bf*)(gwl + kc + 8);
}

__device__ __forceinline__ void stage_commit(const Stage& s,
                                             __bf16* sXH, __bf16* sXL,
                                             __bf16* sWH, __bf16* sWL,
                                             int lrow, int lk) {
  const float e[16] = {s.x0.x, s.x0.y, s.x0.z, s.x0.w,
                       s.x1.x, s.x1.y, s.x1.z, s.x1.w,
                       s.x2.x, s.x2.y, s.x2.z, s.x2.w,
                       s.x3.x, s.x3.y, s.x3.z, s.x3.w};
  v8bf h0, h1, l0, l1;
#pragma unroll
  for (int j = 0; j < 8; ++j) {
    __bf16 h = (__bf16)e[j];
    h0[j] = h; l0[j] = (__bf16)(e[j] - (float)h);
  }
#pragma unroll
  for (int j = 0; j < 8; ++j) {
    __bf16 h = (__bf16)e[8 + j];
    h1[j] = h; l1[j] = (__bf16)(e[8 + j] - (float)h);
  }
  *(v8bf*)&sXH[lrow * LDSTR + lk]     = h0;
  *(v8bf*)&sXH[lrow * LDSTR + lk + 8] = h1;
  *(v8bf*)&sXL[lrow * LDSTR + lk]     = l0;
  *(v8bf*)&sXL[lrow * LDSTR + lk + 8] = l1;
  *(v8bf*)&sWH[lrow * LDSTR + lk]     = s.wh0;
  *(v8bf*)&sWH[lrow * LDSTR + lk + 8] = s.wh1;
  *(v8bf*)&sWL[lrow * LDSTR + lk]     = s.wl0;
  *(v8bf*)&sWL[lrow * LDSTR + lk + 8] = s.wl1;
}

__device__ __forceinline__ void compute_chunk(const __bf16* sXH, const __bf16* sXL,
                                              const __bf16* sWH, const __bf16* sWL,
                                              v8f acc[2][4], int waveM, int waveN,
                                              int fm, int akb, int bkb) {
  // A fragments: CDNA5 16-bit A layout (K 0-7/16-23 vs 8-15/24-31 per half)
  v16bf ah[2], al[2];
#pragma unroll
  for (int im = 0; im < 2; ++im) {
    const int m = waveM + im * 16 + fm;
    ah[im] = cat8(*(const v8bf*)&sXH[m * LDSTR + akb],
                  *(const v8bf*)&sXH[m * LDSTR + akb + 16]);
    al[im] = cat8(*(const v8bf*)&sXL[m * LDSTR + akb],
                  *(const v8bf*)&sXL[m * LDSTR + akb + 16]);
  }
  // B fragments: same K layout as A (K 0-7/16-23 vs 8-15/24-31 per half); W rows are W^T columns
#pragma unroll
  for (int in = 0; in < 4; ++in) {
    const int n = waveN + in * 16 + fm;
    const v16bf bh = cat8(*(const v8bf*)&sWH[n * LDSTR + bkb],
                          *(const v8bf*)&sWH[n * LDSTR + bkb + 16]);
    const v16bf bl = cat8(*(const v8bf*)&sWL[n * LDSTR + bkb],
                          *(const v8bf*)&sWL[n * LDSTR + bkb + 16]);
#pragma unroll
    for (int im = 0; im < 2; ++im) {
      acc[im][in] = __builtin_amdgcn_wmma_f32_16x16x32_bf16(
          false, ah[im], false, bh, (short)0, acc[im][in], false, false);
      acc[im][in] = __builtin_amdgcn_wmma_f32_16x16x32_bf16(
          false, ah[im], false, bl, (short)0, acc[im][in], false, false);
      acc[im][in] = __builtin_amdgcn_wmma_f32_16x16x32_bf16(
          false, al[im], false, bh, (short)0, acc[im][in], false, false);
    }
  }
}

__global__ __launch_bounds__(256) void mpo_gemm(const float*  __restrict__ x,
                                                const __bf16* __restrict__ wh,
                                                const __bf16* __restrict__ wl,
                                                float* __restrict__ y) {
  __shared__ __bf16 sXH[BM * LDSTR];
  __shared__ __bf16 sXL[BM * LDSTR];
  __shared__ __bf16 sWH[BN * LDSTR];
  __shared__ __bf16 sWL[BN * LDSTR];

  const int tid  = threadIdx.x;
  const int bm   = blockIdx.y * BM;
  const int bn   = blockIdx.x * BN;
  const int wid  = tid >> 5;
  const int lane = tid & 31;
  const int waveM = (wid & 3) * 32;     // 4 waves along M
  const int waveN = (wid >> 2) * 64;    // 2 waves along N
  const int fm    = lane & 15;
  const int hi16  = lane >> 4;
  const int akb   = hi16 * 8;
  const int bkb   = hi16 * 8;

  const int lrow = tid >> 1;            // 0..127: tile row this thread stages
  const int lk   = (tid & 1) * 16;      // 0 or 16: k-offset within chunk

  v8f acc[2][4];
#pragma unroll
  for (int im = 0; im < 2; ++im)
#pragma unroll
    for (int in = 0; in < 4; ++in)
#pragma unroll
      for (int r = 0; r < 8; ++r) acc[im][in][r] = 0.0f;

  const float*  gx  = x  + (size_t)(bm + lrow) * IN_F + lk;
  const __bf16* gwh = wh + (size_t)(bn + lrow) * IN_F + lk;
  const __bf16* gwl = wl + (size_t)(bn + lrow) * IN_F + lk;

  Stage s0, s1;
  stage_load(s0, gx, gwh, gwl, 0);

#pragma unroll 1
  for (int kc = 0; kc < IN_F; kc += 2 * BK) {
    // ---- phase A: commit s0, prefetch kc+BK, compute ----
    stage_commit(s0, sXH, sXL, sWH, sWL, lrow, lk);
    __syncthreads();
    stage_load(s1, gx, gwh, gwl, kc + BK);           // hidden under WMMAs
    compute_chunk(sXH, sXL, sWH, sWL, acc, waveM, waveN, fm, akb, bkb);
    __syncthreads();

    // ---- phase B: commit s1, prefetch kc+2*BK (wraps harmlessly), compute ----
    stage_commit(s1, sXH, sXL, sWH, sWL, lrow, lk);
    __syncthreads();
    const int k2 = (kc + 2 * BK < IN_F) ? (kc + 2 * BK) : 0;
    stage_load(s0, gx, gwh, gwl, k2);                // hidden under WMMAs
    compute_chunk(sXH, sXL, sWH, sWL, acc, waveM, waveN, fm, akb, bkb);
    __syncthreads();
  }

  // ---- epilogue: C/D layout (VGPR r: M=r + 8*(lane>=16), N=lane&15) ----
#pragma unroll
  for (int im = 0; im < 2; ++im) {
    const int rowbase = bm + waveM + im * 16 + hi16 * 8;
#pragma unroll
    for (int in = 0; in < 4; ++in) {
      const int col = bn + waveN + in * 16 + fm;
#pragma unroll
      for (int r = 0; r < 8; ++r)
        y[(size_t)(rowbase + r) * OUT_F + col] = acc[im][in][r];
    }
  }
}

// ---------------------------------------------------------------------------
extern "C" void kernel_launch(void* const* d_in, const int* in_sizes, int n_in,
                              void* d_out, int out_size, void* d_ws, size_t ws_size,
                              hipStream_t stream) {
  (void)in_sizes; (void)n_in; (void)out_size; (void)ws_size;
  const float* x     = (const float*)d_in[0];
  const float* core0 = (const float*)d_in[1];
  const float* core1 = (const float*)d_in[2];
  const float* core2 = (const float*)d_in[3];
  float* y = (float*)d_out;

  // workspace layout: T12 (16 MB fp32) | Wh (32 MB bf16) | Wl (32 MB bf16)
  float*  t12 = (float*)d_ws;
  __bf16* wh  = (__bf16*)((char*)d_ws + (size_t)16 * 1024 * 1024);
  __bf16* wl  = wh + (size_t)OUT_F * IN_F;

  k_t12   <<<dim3(64 * 256), dim3(256), 0, stream>>>(core1, core2, t12);
  k_wsplit<<<dim3(256),      dim3(256), 0, stream>>>(core0, t12, wh, wl);
  mpo_gemm<<<dim3(OUT_F / BN, TOKENS / BM), dim3(256), 0, stream>>>(x, wh, wl, y);
}
